// VisitGraphTransformer_70798240907542
// MI455X (gfx1250) — matrix operand order repaired and checked by reading, not compiled, not run on hardware
//
#include <hip/hip_runtime.h>
#include <hip/hip_bf16.h>

typedef __attribute__((ext_vector_type(16))) _Float16 v16h;
typedef __attribute__((ext_vector_type(8)))  float    v8f;

#define NEG_SLOPE 0.2f
#define LN_EPS 1e-5f

__device__ __forceinline__ float wave_sum(float v) {
#pragma unroll
  for (int m = 16; m > 0; m >>= 1) v += __shfl_xor(v, m, 32);
  return v;
}

__device__ __forceinline__ void atomAddF(float* p, float v) {
  unsafeAtomicAdd(p, v);  // -> global_atomic_add_f32 (L2 atomic)
}

// ---------------------------------------------------------------------------
// Zero fill
// ---------------------------------------------------------------------------
__global__ void zero_f32(float* __restrict__ p, long long n) {
  long long i = (long long)blockIdx.x * blockDim.x + threadIdx.x;
  long long stride = (long long)gridDim.x * blockDim.x;
  for (; i < n; i += stride) p[i] = 0.f;
}

// ---------------------------------------------------------------------------
// GEMM: Y[nrows,128] = X[nrows,128] @ W[128,128], split-f16 WMMA (3 products)
// 128 threads = 4 waves; each wave does a 16-row strip, all 128 output cols.
// W is converted to hi/lo f16 B-fragments in LDS using the CDNA5 16-bit
// B-matrix VGPR layout (lane = col + 16*((k>>3)&1), half = (k&7) + 8*(k>=16)).
// A fragments built from global per the 16-bit A layout
// (lanes 0-15: K = {0..7, 16..23}; lanes 16-31: +8).
// ---------------------------------------------------------------------------
__global__ __launch_bounds__(128) void gemm128_wmma(
    const float* __restrict__ X, const float* __restrict__ W,
    float* __restrict__ Y, int nrows) {
  __shared__ __attribute__((aligned(32))) _Float16 sBhi[32 * 32 * 16];  // 32 frags * 32 lanes * 16 halves
  __shared__ __attribute__((aligned(32))) _Float16 sBlo[32 * 32 * 16];

  int tid = threadIdx.x;
  for (int idx = tid; idx < 128 * 128; idx += 128) {
    int k = idx >> 7, n = idx & 127;
    float w = W[idx];
    _Float16 hi = (_Float16)w;
    _Float16 lo = (_Float16)(w - (float)hi);
    int ks = k >> 5, kk = k & 31;
    int nt = n >> 4, nn = n & 15;
    int lane = nn + ((kk & 8) ? 16 : 0);
    int h = (kk & 7) | ((kk & 16) >> 1);
    int dst = (((nt << 2) | ks) * 32 + lane) * 16 + h;
    sBhi[dst] = hi;
    sBlo[dst] = lo;
  }
  __syncthreads();

  int wave = tid >> 5, lane = tid & 31;
  int rowBase = blockIdx.x * 64 + wave * 16;
  if (rowBase >= nrows) return;

  const float* xr = X + (size_t)(rowBase + (lane & 15)) * 128;
  int half = (lane >> 4) << 3;  // +8 for upper half-lanes

  v8f zero8 = {0.f, 0.f, 0.f, 0.f, 0.f, 0.f, 0.f, 0.f};
  v8f acc[8];
#pragma unroll
  for (int t = 0; t < 8; t++) acc[t] = zero8;

#pragma unroll
  for (int ks = 0; ks < 4; ks++) {
    int kb = ks * 32 + half;
    float f[16];
#pragma unroll
    for (int j = 0; j < 8; j++) f[j] = xr[kb + j];
#pragma unroll
    for (int j = 0; j < 8; j++) f[8 + j] = xr[kb + 16 + j];
    v16h ahi, alo;
#pragma unroll
    for (int j = 0; j < 16; j++) {
      _Float16 hi = (_Float16)f[j];
      ahi[j] = hi;
      alo[j] = (_Float16)(f[j] - (float)hi);
    }
#pragma unroll
    for (int nt = 0; nt < 8; nt++) {
      const v16h bhi = *(const v16h*)&sBhi[(((nt << 2) | ks) * 32 + lane) * 16];
      const v16h blo = *(const v16h*)&sBlo[(((nt << 2) | ks) * 32 + lane) * 16];
      acc[nt] = __builtin_amdgcn_wmma_f32_16x16x32_f16(false, ahi, false, bhi,
                                                       (short)0, acc[nt], false, false);
      acc[nt] = __builtin_amdgcn_wmma_f32_16x16x32_f16(false, ahi, false, blo,
                                                       (short)0, acc[nt], false, false);
      acc[nt] = __builtin_amdgcn_wmma_f32_16x16x32_f16(false, alo, false, bhi,
                                                       (short)0, acc[nt], false, false);
    }
  }

  // D layout: VGPR r -> row r (lanes 0-15) / row 8+r (lanes 16-31), col = lane&15
  int rowOut = rowBase + ((lane >> 4) << 3);
  int coln = lane & 15;
#pragma unroll
  for (int nt = 0; nt < 8; nt++) {
    float* yp = Y + (size_t)rowOut * 128 + nt * 16 + coln;
#pragma unroll
    for (int r = 0; r < 8; r++) yp[(size_t)r * 128] = acc[nt][r];
  }
}

// ---------------------------------------------------------------------------
// Per-node attention logits: a_src[n,h] = dot(h[n, h*C:(h+1)*C], attW[h]).
// One wave per node; head h covers channels [32h, 32h+32) for HEADS=4,
// all 128 channels for HEADS=1.
// ---------------------------------------------------------------------------
template <int HEADS>
__global__ void attdot_kernel(const float* __restrict__ h,
                              const float* __restrict__ asrcW,
                              const float* __restrict__ adstW,
                              float* __restrict__ asrc, float* __restrict__ adst,
                              int n) {
  int node = blockIdx.x * (blockDim.x >> 5) + (threadIdx.x >> 5);
  if (node >= n) return;
  int lane = threadIdx.x & 31;
  const float* hr = h + (size_t)node * 128;
  float sv[4], dv[4];
#pragma unroll
  for (int i = 0; i < 4; i++) {
    float v = hr[i * 32 + lane];
    sv[i] = v * asrcW[i * 32 + lane];
    dv[i] = v * adstW[i * 32 + lane];
  }
  if (HEADS == 4) {
#pragma unroll
    for (int i = 0; i < 4; i++) {
      float s = wave_sum(sv[i]);
      float d = wave_sum(dv[i]);
      if (lane == 0) {
        asrc[(size_t)node * 4 + i] = s;
        adst[(size_t)node * 4 + i] = d;
      }
    }
  } else {
    float s = wave_sum(sv[0] + sv[1] + sv[2] + sv[3]);
    float d = wave_sum(dv[0] + dv[1] + dv[2] + dv[3]);
    if (lane == 0) {
      asrc[node] = s;
      adst[node] = d;
    }
  }
}

// ---------------------------------------------------------------------------
// Edge alpha pass: e = exp(leakyrelu(asrc[s]+adst[d])), denom[d] += e.
// (segment-max skipped: softmax is shift-invariant and logits are O(1))
// ---------------------------------------------------------------------------
__global__ void alpha4_kernel(const float* __restrict__ asrc,
                              const float* __restrict__ adst,
                              const int* __restrict__ ei, int E, int Etot,
                              float* __restrict__ ebuf, float* __restrict__ den) {
  int idx = blockIdx.x * blockDim.x + threadIdx.x;
  if (idx >= Etot * 4) return;
  int e = idx >> 2, hd = idx & 3;
  int s, d;
  if (e < E) { s = ei[e]; d = ei[E + e]; } else { s = d = e - E; }
  float a = asrc[(size_t)s * 4 + hd] + adst[(size_t)d * 4 + hd];
  a = (a > 0.f) ? a : NEG_SLOPE * a;
  float ex = __expf(a);
  ebuf[idx] = ex;
  atomAddF(&den[(size_t)d * 4 + hd], ex);
}

__global__ void alpha1_kernel(const float* __restrict__ asrc,
                              const float* __restrict__ adst,
                              const int* __restrict__ ei, int E, int Etot,
                              float* __restrict__ ebuf, float* __restrict__ den) {
  int e = blockIdx.x * blockDim.x + threadIdx.x;
  if (e >= Etot) return;
  int s, d;
  if (e < E) { s = ei[e]; d = ei[E + e]; } else { s = d = e - E; }
  float a = asrc[s] + adst[d];
  a = (a > 0.f) ? a : NEG_SLOPE * a;
  float ex = __expf(a);
  ebuf[e] = ex;
  atomAddF(&den[d], ex);
}

// ---------------------------------------------------------------------------
// Message scatter: agg[d] += h[s] * coef. One wave per edge, 4 floats/lane.
// ---------------------------------------------------------------------------
__global__ void msg4_kernel(const float* __restrict__ h,
                            const float* __restrict__ ebuf,
                            const float* __restrict__ den,
                            const int* __restrict__ ei, int E, int Etot,
                            float* __restrict__ agg) {
  int e = blockIdx.x * (blockDim.x >> 5) + (threadIdx.x >> 5);
  if (e >= Etot) return;
  int lane = threadIdx.x & 31;
  int s, d;
  if (e < E) { s = ei[e]; d = ei[E + e]; } else { s = d = e - E; }
  const float* hs = h + (size_t)s * 128;
  float* od = agg + (size_t)d * 128;
#pragma unroll
  for (int i = 0; i < 4; i++) {
    float coef = ebuf[(size_t)e * 4 + i] / den[(size_t)d * 4 + i];
    int c = i * 32 + lane;
    atomAddF(&od[c], hs[c] * coef);
  }
}

__global__ void msg1_kernel(const float* __restrict__ h,
                            const float* __restrict__ ebuf,
                            const float* __restrict__ den,
                            const int* __restrict__ ei, int E, int Etot,
                            float* __restrict__ agg) {
  int e = blockIdx.x * (blockDim.x >> 5) + (threadIdx.x >> 5);
  if (e >= Etot) return;
  int lane = threadIdx.x & 31;
  int s, d;
  if (e < E) { s = ei[e]; d = ei[E + e]; } else { s = d = e - E; }
  float coef = ebuf[e] / den[d];
  const float* hs = h + (size_t)s * 128;
  float* od = agg + (size_t)d * 128;
#pragma unroll
  for (int i = 0; i < 4; i++) {
    int c = i * 32 + lane;
    atomAddF(&od[c], hs[c] * coef);
  }
}

// ---------------------------------------------------------------------------
// Layer-1 epilogue: h1 = relu(agg + b1) in place.
// ---------------------------------------------------------------------------
__global__ void relu_bias_kernel(float* __restrict__ buf,
                                 const float* __restrict__ bias, long long n) {
  long long i = (long long)blockIdx.x * blockDim.x + threadIdx.x;
  if (i >= n) return;
  float v = buf[i] + bias[i & 127];
  buf[i] = v > 0.f ? v : 0.f;
}

// ---------------------------------------------------------------------------
// LayerNorm + per-graph mean-pool accumulation. One wave per node.
// ---------------------------------------------------------------------------
__global__ void ln_pool_kernel(const float* __restrict__ agg,
                               const float* __restrict__ b2,
                               const float* __restrict__ gamma,
                               const float* __restrict__ beta,
                               const int* __restrict__ batch,
                               float* __restrict__ out, float* __restrict__ cnt,
                               int n) {
  int node = blockIdx.x * (blockDim.x >> 5) + (threadIdx.x >> 5);
  if (node >= n) return;
  int lane = threadIdx.x & 31;
  const float* ar = agg + (size_t)node * 128;
  float v[4];
#pragma unroll
  for (int i = 0; i < 4; i++) v[i] = ar[i * 32 + lane] + b2[i * 32 + lane];
  float mean = wave_sum(v[0] + v[1] + v[2] + v[3]) * (1.f / 128.f);
  float sq = 0.f;
#pragma unroll
  for (int i = 0; i < 4; i++) {
    float t = v[i] - mean;
    sq += t * t;
  }
  float var = wave_sum(sq) * (1.f / 128.f);
  float rstd = rsqrtf(var + LN_EPS);
  int g = batch[node];
  float* og = out + (size_t)g * 128;
#pragma unroll
  for (int i = 0; i < 4; i++) {
    int c = i * 32 + lane;
    float y = (v[i] - mean) * rstd * gamma[c] + beta[c];
    atomAddF(&og[c], y);
  }
  if (lane == 0) atomAddF(&cnt[g], 1.f);
}

__global__ void finalize_kernel(float* __restrict__ out,
                                const float* __restrict__ cnt, long long n) {
  long long i = (long long)blockIdx.x * blockDim.x + threadIdx.x;
  if (i >= n) return;
  out[i] = out[i] / fmaxf(cnt[i >> 7], 1.0f);
}

// ---------------------------------------------------------------------------
extern "C" void kernel_launch(void* const* d_in, const int* in_sizes, int n_in,
                              void* d_out, int out_size, void* d_ws, size_t ws_size,
                              hipStream_t stream) {
  const float* x       = (const float*)d_in[0];
  const int*   ei      = (const int*)d_in[1];
  const int*   batch   = (const int*)d_in[2];
  const float* W1      = (const float*)d_in[3];
  const float* attS1   = (const float*)d_in[4];
  const float* attD1   = (const float*)d_in[5];
  const float* b1      = (const float*)d_in[6];
  const float* W2      = (const float*)d_in[7];
  const float* attS2   = (const float*)d_in[8];
  const float* attD2   = (const float*)d_in[9];
  const float* b2      = (const float*)d_in[10];
  const float* gamma   = (const float*)d_in[11];
  const float* beta    = (const float*)d_in[12];

  const int N = in_sizes[0] / 128;
  const int E = in_sizes[1] / 2;
  const int Etot = E + N;          // edges + self loops
  const int G = out_size / 128;

  float* ws = (float*)d_ws;
  size_t o = 0;
  float* bufA  = ws + o; o += (size_t)N * 128;  // h_lin1, later h_lin2
  float* bufB  = ws + o; o += (size_t)N * 128;  // agg1/h1, later agg2
  float* asrc1 = ws + o; o += (size_t)N * 4;
  float* adst1 = ws + o; o += (size_t)N * 4;
  float* ebuf1 = ws + o; o += (size_t)Etot * 4;
  float* den1  = ws + o; o += (size_t)N * 4;
  float* asrc2 = ws + o; o += (size_t)N;
  float* adst2 = ws + o; o += (size_t)N;
  float* ebuf2 = ws + o; o += (size_t)Etot;
  float* den2  = ws + o; o += (size_t)N;
  float* cnt   = ws + o; o += (size_t)G;
  float* outF  = (float*)d_out;

  const int ZB = 2048;
  // init accumulators
  zero_f32<<<ZB, 256, 0, stream>>>(bufB, (long long)N * 128);
  zero_f32<<<ZB, 256, 0, stream>>>(den1, (long long)N * 4);
  zero_f32<<<ZB, 256, 0, stream>>>(den2, (long long)N);
  zero_f32<<<ZB, 256, 0, stream>>>(outF, (long long)G * 128);
  zero_f32<<<ZB, 256, 0, stream>>>(cnt, (long long)G);

  int gemmBlocks = (N + 63) / 64;
  int nodeWaveBlocks = (N + 7) / 8;      // 8 waves (nodes) per 256-thread block
  int edgeWaveBlocks = (Etot + 7) / 8;

  // ---- Layer 1 ----
  gemm128_wmma<<<gemmBlocks, 128, 0, stream>>>(x, W1, bufA, N);
  attdot_kernel<4><<<nodeWaveBlocks, 256, 0, stream>>>(bufA, attS1, attD1,
                                                       asrc1, adst1, N);
  alpha4_kernel<<<(Etot * 4 + 255) / 256, 256, 0, stream>>>(asrc1, adst1, ei, E,
                                                            Etot, ebuf1, den1);
  msg4_kernel<<<edgeWaveBlocks, 256, 0, stream>>>(bufA, ebuf1, den1, ei, E,
                                                  Etot, bufB);
  relu_bias_kernel<<<(int)(((long long)N * 128 + 255) / 256), 256, 0, stream>>>(
      bufB, b1, (long long)N * 128);

  // ---- Layer 2 ----
  gemm128_wmma<<<gemmBlocks, 128, 0, stream>>>(bufB, W2, bufA, N);
  attdot_kernel<1><<<nodeWaveBlocks, 256, 0, stream>>>(bufA, attS2, attD2,
                                                       asrc2, adst2, N);
  zero_f32<<<ZB, 256, 0, stream>>>(bufB, (long long)N * 128);  // now agg2
  alpha1_kernel<<<(Etot + 255) / 256, 256, 0, stream>>>(asrc2, adst2, ei, E,
                                                        Etot, ebuf2, den2);
  msg1_kernel<<<edgeWaveBlocks, 256, 0, stream>>>(bufA, ebuf2, den2, ei, E,
                                                  Etot, bufB);

  // ---- LayerNorm + pool ----
  ln_pool_kernel<<<nodeWaveBlocks, 256, 0, stream>>>(bufB, b2, gamma, beta,
                                                     batch, outF, cnt, N);
  finalize_kernel<<<(int)(((long long)G * 128 + 255) / 256), 256, 0, stream>>>(
      outF, cnt, (long long)G * 128);
}
